// FastRCNNOutputLayers_66451734003796
// MI455X (gfx1250) — matrix operand order repaired and checked by reading, not compiled, not run on hardware
//
#include <hip/hip_runtime.h>
#include <cstdint>

// ---------------- problem constants (from reference) ----------------
#define NROW   8192          // N
#define DDIM   1024          // D
#define KCLS   80            // K
#define NCLS   81            // K+1 (softmax incl. background)
#define NBOX   320           // K*4
#define NCOL   401           // 81 + 320 combined GEMM output cols
#define NCOLP  416           // padded to 26 * 16
#define NTILEN 26
#define NTILEM 512           // 8192 / 16
#define NK     (NROW * KCLS) // 655360 candidates
#define MCAND  2048          // pre-NMS budget
#define CAP    8192          // compaction cap (pow2, sorted in LDS)
#define TOPK   100

__constant__ const float SCORE_T     = 0.05f;
__constant__ const float NMS_T       = 0.5f;
__constant__ const float IMG_W       = 1216.0f;
__constant__ const float IMG_H       = 800.0f;
__constant__ const float MAX_COORD   = 1217.0f;              // max(H,W)+1
__constant__ const float SCALE_CLAMP = 4.135166556742356f;   // log(1000/16)

typedef __bf16 bf16_t;
typedef __attribute__((ext_vector_type(8)))  bf16_t v8bf;   // 16 bytes -> one b128
typedef __attribute__((ext_vector_type(16))) bf16_t v16bf;  // WMMA A/B operand
typedef __attribute__((ext_vector_type(8)))  float  v8f;    // WMMA C/D operand

// float -> bf16, round-to-nearest-even (no reliance on cast support)
__device__ __forceinline__ bf16_t f2bf(float f) {
  union { float f; unsigned u; } in; in.f = f;
  unsigned u = in.u;
  unsigned r = u + 0x7FFFu + ((u >> 16) & 1u);
  union { unsigned short s; bf16_t b; } out;
  out.s = (unsigned short)(r >> 16);
  return out.b;
}

// ---------------- 0a) convert activations to bf16 row-major [8192][1024]
__global__ void pack_x(const float* __restrict__ x, bf16_t* __restrict__ xbf) {
  int i = blockIdx.x * blockDim.x + threadIdx.x;   // over NROW*DDIM/4
  if (i >= (NROW * DDIM) / 4) return;
  const float4 v = ((const float4*)x)[i];
  bf16_t* o = xbf + (size_t)i * 4;
  o[0] = f2bf(v.x); o[1] = f2bf(v.y); o[2] = f2bf(v.z); o[3] = f2bf(v.w);
}

// ---------------- 0b) pack weights: [1024,81]+[1024,320] -> col-major bf16 [416][1024]
__global__ void pack_weights(const float* __restrict__ cls_w,
                             const float* __restrict__ bbox_w,
                             bf16_t* __restrict__ wbf) {
  int i = blockIdx.x * blockDim.x + threadIdx.x;
  if (i >= NCOLP * DDIM) return;
  int c = i >> 10;           // column (0..415)
  int k = i & (DDIM - 1);    // depth
  float v = 0.0f;
  if (c < NCLS)       v = cls_w[(size_t)k * NCLS + c];
  else if (c < NCOL)  v = bbox_w[(size_t)k * NBOX + (c - NCLS)];
  wbf[(size_t)c * DDIM + k] = f2bf(v);
}

// ---------------- 1) GEMM: logits[8192,416] = x[8192,1024] * W[1024,416]
// One wave per (16-row tile x 2x16-col tiles). Steady-state inner loop:
// 2 A b128 loads + 4 B b128 loads + 2 v_wmma_f32_16x16x32_bf16.
__global__ __launch_bounds__(32)
void gemm_wmma_bf16(const bf16_t* __restrict__ xbf,
                    const bf16_t* __restrict__ wbf,
                    float* __restrict__ logits) {
  const int tm   = blockIdx.x;        // 0..511 row tile
  const int tg   = blockIdx.y;        // 0..12  col-tile pair
  const int lane = threadIdx.x;       // 0..31 (wave32)
  const int half = lane >> 4;         // 0/1
  const int l15  = lane & 15;

  const int row  = tm * 16 + l15;             // A row held by this lane
  const int col0 = (tg * 2) * 16 + l15;       // B column, tile 0
  const int col1 = col0 + 16;                 // B column, tile 1

  const bf16_t* arow  = xbf + (size_t)row * DDIM;
  const bf16_t* bcol0 = wbf + (size_t)col0 * DDIM + half * 16;
  const bf16_t* bcol1 = wbf + (size_t)col1 * DDIM + half * 16;

  v8f acc0 = {};
  v8f acc1 = {};
  union AU { v8bf h[2]; v16bf v; };

  for (int k0 = 0; k0 < DDIM; k0 += 32) {
    if (k0 + 32 < DDIM)
      __builtin_prefetch(arow + k0 + 32, 0, 0);        // global_prefetch_b8

    // A layout: lane m=l15; elems 0..15 -> K=k0+half*16+e (matches B).
    // Packed bf16 -> two b128 loads.
    AU a;
    a.h[0] = *(const v8bf*)(arow + k0 + half * 16);
    a.h[1] = *(const v8bf*)(arow + k0 + half * 16 + 8);

    // B layout: lane n=l15, K = k0 + half*16 + e (contiguous in packed buffer)
    v16bf b0 = *(const v16bf*)(bcol0 + k0);
    v16bf b1 = *(const v16bf*)(bcol1 + k0);

    acc0 = __builtin_amdgcn_wmma_f32_16x16x32_bf16(
        false, a.v, false, b0, (short)0, acc0, false, false);
    acc1 = __builtin_amdgcn_wmma_f32_16x16x32_bf16(
        false, a.v, false, b1, (short)0, acc1, false, false);
  }

  // C/D layout: lane col = l15; VGPR r -> global row = tm*16 + half*8 + r
  float* out0 = logits + (size_t)(tm * 16 + half * 8) * NCOLP + col0;
#pragma unroll
  for (int r = 0; r < 8; ++r) out0[(size_t)r * NCOLP] = acc0[r];
  float* out1 = out0 + 16;
#pragma unroll
  for (int r = 0; r < 8; ++r) out1[(size_t)r * NCOLP] = acc1[r];
}

// ---------------- 2) per-row softmax + box decode/clip
__global__ __launch_bounds__(128)
void postprocess(const float* __restrict__ logits,
                 const float* __restrict__ cls_b,
                 const float* __restrict__ bbox_b,
                 const float* __restrict__ proposals,
                 float* __restrict__ probs,
                 float* __restrict__ boxes) {
  const int row = blockIdx.x;
  const int t   = threadIdx.x;
  __shared__ float red[128];
  const float* lrow = logits + (size_t)row * NCOLP;

  float l = -3.4e38f;
  if (t < NCLS) l = lrow[t] + cls_b[t];
  red[t] = l;
  __syncthreads();
  for (int s = 64; s > 0; s >>= 1) { if (t < s) red[t] = fmaxf(red[t], red[t + s]); __syncthreads(); }
  float mx = red[0];
  __syncthreads();
  float e = (t < NCLS) ? __expf(l - mx) : 0.0f;
  red[t] = e;
  __syncthreads();
  for (int s = 64; s > 0; s >>= 1) { if (t < s) red[t] += red[t + s]; __syncthreads(); }
  float inv = 1.0f / red[0];

  if (t < KCLS) {
    probs[(size_t)row * KCLS + t] = e * inv;

    float x1 = proposals[row * 4 + 0], y1 = proposals[row * 4 + 1];
    float x2 = proposals[row * 4 + 2], y2 = proposals[row * 4 + 3];
    float w = x2 - x1, h = y2 - y1;
    float cx = x1 + 0.5f * w, cy = y1 + 0.5f * h;

    const float* d = lrow + NCLS + t * 4;
    float dx = (d[0] + bbox_b[t * 4 + 0]) * 0.1f;
    float dy = (d[1] + bbox_b[t * 4 + 1]) * 0.1f;
    float dw = fminf((d[2] + bbox_b[t * 4 + 2]) * 0.2f, SCALE_CLAMP);
    float dh = fminf((d[3] + bbox_b[t * 4 + 3]) * 0.2f, SCALE_CLAMP);

    float pcx = dx * w + cx, pcy = dy * h + cy;
    float pw = __expf(dw) * w, ph = __expf(dh) * h;
    float bx1 = fminf(fmaxf(pcx - 0.5f * pw, 0.0f), IMG_W);
    float by1 = fminf(fmaxf(pcy - 0.5f * ph, 0.0f), IMG_H);
    float bx2 = fminf(fmaxf(pcx + 0.5f * pw, 0.0f), IMG_W);
    float by2 = fminf(fmaxf(pcy + 0.5f * ph, 0.0f), IMG_H);
    float* bo = boxes + ((size_t)row * KCLS + t) * 4;
    bo[0] = bx1; bo[1] = by1; bo[2] = bx2; bo[3] = by2;
  }
}

// ---------------- 3) top-2048 selection: histogram pivot + compaction + LDS bitonic sort
__global__ void zero_init(int* hist, int* counter, int* pivot,
                          unsigned long long* pairs) {
  int i = blockIdx.x * blockDim.x + threadIdx.x;
  if (i < 1024) hist[i] = 0;
  if (i == 0) { *counter = 0; *pivot = 51; }
  if (i < CAP) pairs[i] = 0ull;
}

__global__ void score_hist(const float* __restrict__ probs, int* __restrict__ hist) {
  int i = blockIdx.x * blockDim.x + threadIdx.x;
  if (i >= NK) return;
  float s = probs[i];
  if (s > SCORE_T) {
    int b = (int)(s * 1024.0f); if (b > 1023) b = 1023;
    atomicAdd(&hist[b], 1);
  }
}

__global__ void find_pivot(const int* __restrict__ hist, int* __restrict__ pivot) {
  if (blockIdx.x == 0 && threadIdx.x == 0) {
    int acc = 0;
    for (int b = 1023; b >= 0; --b) {
      acc += hist[b];
      if (acc >= MCAND) { *pivot = b; return; }
    }
  }
}

__global__ void compact_cands(const float* __restrict__ probs,
                              const int* __restrict__ pivot,
                              int* __restrict__ counter,
                              unsigned long long* __restrict__ pairs) {
  int i = blockIdx.x * blockDim.x + threadIdx.x;
  if (i >= NK) return;
  float s = probs[i];
  if (s > SCORE_T) {
    int b = (int)(s * 1024.0f); if (b > 1023) b = 1023;
    if (b >= *pivot) {
      int pos = atomicAdd(counter, 1);
      if (pos < CAP)
        pairs[pos] = ((unsigned long long)__float_as_uint(s) << 32) |
                     (unsigned)(~(unsigned)i);   // ~idx: ties -> lower idx first (desc sort)
    }
  }
}

// descending bitonic sort in LDS (key: u64; zero keys sink to the end)
__device__ __forceinline__ void bitonic_desc(unsigned long long* buf, int n,
                                             int tid, int nthreads) {
  for (int k = 2; k <= n; k <<= 1) {
    for (int j = k >> 1; j > 0; j >>= 1) {
      for (int i = tid; i < n; i += nthreads) {
        int partner = i ^ j;
        if (partner > i) {
          unsigned long long a = buf[i], b = buf[partner];
          bool up = ((i & k) == 0);
          bool sw = up ? (a < b) : (a > b);
          if (sw) { buf[i] = b; buf[partner] = a; }
        }
      }
      __syncthreads();
    }
  }
}

__global__ __launch_bounds__(1024)
void sort_topM(const unsigned long long* __restrict__ pairs,
               float* __restrict__ top_scores, int* __restrict__ top_idx) {
  __shared__ unsigned long long sb[CAP];  // 64 KB of the 320 KB WGP LDS
  const int t = threadIdx.x;
  for (int i = t; i < CAP; i += 1024) sb[i] = pairs[i];
  __syncthreads();
  bitonic_desc(sb, CAP, t, 1024);
  for (int i = t; i < MCAND; i += 1024) {
    unsigned long long key = sb[i];
    unsigned fb = (unsigned)(key >> 32);
    if (fb) { top_scores[i] = __uint_as_float(fb); top_idx[i] = (int)(~(unsigned)key); }
    else    { top_scores[i] = -1.0f;               top_idx[i] = 0; }
  }
}

// ---------------- 4) gather candidate boxes (+ per-class offset for batched NMS)
__global__ void gather_cands(const float* __restrict__ boxes,
                             const float* __restrict__ top_scores,
                             const int* __restrict__ top_idx,
                             float4* __restrict__ cand_raw,
                             float4* __restrict__ cand_off) {
  int m = blockIdx.x * blockDim.x + threadIdx.x;
  if (m >= MCAND) return;
  int idx = top_idx[m];
  const float* b = boxes + (size_t)idx * 4;
  float4 raw; raw.x = b[0]; raw.y = b[1]; raw.z = b[2]; raw.w = b[3];
  cand_raw[m] = raw;
  float off = (float)(idx % KCLS) * MAX_COORD;
  float4 o; o.x = raw.x + off; o.y = raw.y + off; o.z = raw.z + off; o.w = raw.w + off;
  cand_off[m] = o;
}

__device__ __forceinline__ float iou4(float4 a, float4 b) {
  float x1 = fmaxf(a.x, b.x), y1 = fmaxf(a.y, b.y);
  float x2 = fminf(a.z, b.z), y2 = fminf(a.w, b.w);
  float inter = fmaxf(x2 - x1, 0.0f) * fmaxf(y2 - y1, 0.0f);
  float aa = (a.z - a.x) * (a.w - a.y);
  float ab = (b.z - b.x) * (b.w - b.y);
  return inter / fmaxf(aa + ab - inter, 1e-9f);
}

// ---------------- 5) bitmask NMS: mask[i][cword] has bit jj set iff j>i && IoU>thr
__global__ __launch_bounds__(64)
void nms_mask(const float4* __restrict__ cand_off,
              unsigned long long* __restrict__ mask) {
  const int ct = blockIdx.x;   // col tile (0..31), 64 j's
  const int rt = blockIdx.y;   // row tile (0..31), 64 i's
  const int t  = threadIdx.x;
  __shared__ float4 jb[64];
  jb[t] = cand_off[ct * 64 + t];
  __syncthreads();
  const int i = rt * 64 + t;
  const float4 ib = cand_off[i];
  unsigned long long w = 0ull;
#pragma unroll 4
  for (int jj = 0; jj < 64; ++jj) {
    int j = ct * 64 + jj;
    if (j > i && iou4(ib, jb[jj]) > NMS_T) w |= (1ull << jj);
  }
  mask[(size_t)i * 32 + ct] = w;
}

// serial score-ordered reduction on one wave32: lane l owns j in [64l, 64l+64)
__global__ __launch_bounds__(32)
void nms_reduce(const unsigned long long* __restrict__ mask,
                const float* __restrict__ top_scores,
                unsigned long long* __restrict__ keep_words) {
  const int lane = threadIdx.x;
  unsigned long long validw = 0ull;
  for (int b = 0; b < 64; ++b)
    if (top_scores[lane * 64 + b] > SCORE_T) validw |= (1ull << b);

  unsigned long long supp = 0ull;
  unsigned long long rowv = mask[(size_t)0 * 32 + lane];  // prefetch row 0
  for (int i = 0; i < MCAND; ++i) {
    unsigned long long cur = rowv;
    if (i + 1 < MCAND) rowv = mask[(size_t)(i + 1) * 32 + lane];  // hide latency
    int owner = i >> 6, bit = i & 63;
    int sbit = (int)((supp   >> bit) & 1ull);
    int vbit = (int)((validw >> bit) & 1ull);
    int s_i = __shfl(sbit, owner, 32);
    int v_i = __shfl(vbit, owner, 32);
    if (!s_i && v_i) supp |= cur;   // i alive -> suppress its higher-index overlaps
  }
  keep_words[lane] = validw & ~supp;
}

// ---------------- 6) final top-100 over kept scores, gather outputs
__global__ __launch_bounds__(1024)
void final_topk(const float* __restrict__ top_scores,
                const int* __restrict__ top_idx,
                const float4* __restrict__ cand_raw,
                const unsigned long long* __restrict__ keep_words,
                float* __restrict__ out) {
  __shared__ unsigned long long sb[MCAND];  // 16 KB
  const int t = threadIdx.x;
  for (int i = t; i < MCAND; i += 1024) {
    int kb = (int)((keep_words[i >> 6] >> (i & 63)) & 1ull);
    unsigned long long key;
    if (kb) key = ((unsigned long long)__float_as_uint(top_scores[i]) << 32) |
                  (unsigned)(~(unsigned)i);
    else    key = (unsigned long long)(unsigned)(~(unsigned)i);  // sinks; lower i first
    sb[i] = key;
  }
  __syncthreads();
  bitonic_desc(sb, MCAND, t, 1024);
  if (t < TOPK) {
    unsigned long long key = sb[t];
    unsigned fb = (unsigned)(key >> 32);
    int m = (int)(~(unsigned)key);
    bool kept = (fb != 0);
    float sc = kept ? __uint_as_float(fb) : -1.0f;
    float4 bx = cand_raw[m];
    int cls = top_idx[m] % KCLS;
    out[t * 4 + 0] = bx.x;
    out[t * 4 + 1] = bx.y;
    out[t * 4 + 2] = bx.z;
    out[t * 4 + 3] = bx.w;
    out[4 * TOPK + t]     = sc;                 // fscores
    out[5 * TOPK + t]     = (float)cls;         // fcls
    out[6 * TOPK + t]     = kept ? 1.0f : 0.0f; // fvalid
  }
}

// ---------------- host-side launcher ----------------
extern "C" void kernel_launch(void* const* d_in, const int* in_sizes, int n_in,
                              void* d_out, int out_size, void* d_ws, size_t ws_size,
                              hipStream_t stream) {
  (void)in_sizes; (void)n_in; (void)out_size; (void)ws_size;
  const float* x         = (const float*)d_in[0];
  const float* cls_w     = (const float*)d_in[1];
  const float* cls_b     = (const float*)d_in[2];
  const float* bbox_w    = (const float*)d_in[3];
  const float* bbox_b    = (const float*)d_in[4];
  const float* proposals = (const float*)d_in[5];
  float* out = (float*)d_out;

  char* p = (char*)d_ws;
  auto alloc = [&](size_t bytes) -> char* {
    char* r = p; p += (bytes + 255) & ~(size_t)255; return r;
  };
  bf16_t* xbf      = (bf16_t*)alloc((size_t)NROW * DDIM * sizeof(bf16_t));    // 16.8 MB
  bf16_t* wbf      = (bf16_t*)alloc((size_t)NCOLP * DDIM * sizeof(bf16_t));   // 0.85 MB
  float*  logits   = (float*) alloc((size_t)NROW * NCOLP * sizeof(float));    // 13.6 MB
  float*  probs    = (float*) alloc((size_t)NK * sizeof(float));              // 2.6 MB
  float*  boxes    = (float*) alloc((size_t)NK * 4 * sizeof(float));          // 10.5 MB
  int*    hist     = (int*)   alloc(1024 * sizeof(int));
  int*    counter  = (int*)   alloc(256);
  int*    pivot    = (int*)   alloc(256);
  unsigned long long* pairs = (unsigned long long*)alloc((size_t)CAP * 8);    // 64 KB
  float*  top_sc   = (float*) alloc(MCAND * sizeof(float));
  int*    top_ix   = (int*)   alloc(MCAND * sizeof(int));
  float4* cand_raw = (float4*)alloc(MCAND * sizeof(float4));
  float4* cand_off = (float4*)alloc(MCAND * sizeof(float4));
  unsigned long long* mask = (unsigned long long*)alloc((size_t)MCAND * 32 * 8); // 512 KB
  unsigned long long* keepw = (unsigned long long*)alloc(32 * 8);

  // 0) pack activations + weights to bf16
  pack_x<<<((NROW * DDIM / 4) + 255) / 256, 256, 0, stream>>>(x, xbf);
  pack_weights<<<(NCOLP * DDIM + 255) / 256, 256, 0, stream>>>(cls_w, bbox_w, wbf);

  // 1) fused GEMM: one wave per 16x32 output tile pair, WMMA bf16, K=1024
  gemm_wmma_bf16<<<dim3(NTILEM, NTILEN / 2), 32, 0, stream>>>(xbf, wbf, logits);

  // 2) softmax + box decode
  postprocess<<<NROW, 128, 0, stream>>>(logits, cls_b, bbox_b, proposals, probs, boxes);

  // 3) top-2048 (sorted desc, low-index tiebreak == lax.top_k)
  zero_init<<<(CAP + 255) / 256, 256, 0, stream>>>(hist, counter, pivot, pairs);
  score_hist<<<(NK + 255) / 256, 256, 0, stream>>>(probs, hist);
  find_pivot<<<1, 32, 0, stream>>>(hist, pivot);
  compact_cands<<<(NK + 255) / 256, 256, 0, stream>>>(probs, pivot, counter, pairs);
  sort_topM<<<1, 1024, 0, stream>>>(pairs, top_sc, top_ix);

  // 4) gather candidate boxes
  gather_cands<<<(MCAND + 255) / 256, 256, 0, stream>>>(boxes, top_sc, top_ix,
                                                        cand_raw, cand_off);

  // 5) NMS: parallel bitmask + single-wave ordered reduce
  nms_mask<<<dim3(32, 32), 64, 0, stream>>>(cand_off, mask);
  nms_reduce<<<1, 32, 0, stream>>>(mask, top_sc, keepw);

  // 6) final top-100 + output write
  final_topk<<<1, 1024, 0, stream>>>(top_sc, top_ix, cand_raw, keepw, out);
}
